// KAN_20899310863172
// MI455X (gfx1250) — matrix operand order repaired and checked by reading, not compiled, not run on hardware
//
#include <hip/hip_runtime.h>
#include <hip/hip_bf16.h>
#include <stdint.h>

// ---------------------------------------------------------------------------
// KAN (2 layers) as bf16 WMMA GEMMs on gfx1250.
//   Featurize x -> F0 (9 feats/channel: 8 bspline + silu), pack weights into
//   WMMA B-fragment order, then two GEMMs. Layer-0 epilogue emits layer-1
//   features directly (no h1 buffer). B tiles double-buffered in LDS via
//   global_load_async_to_lds_b128 (ASYNCcnt), A frags straight from L2.
//   Wave tile 32x128 (16 WMMAs / k-tile); WG tile 256x128; last k-iteration
//   peeled so the hot loop is branch-free. FEAT epilogue stages C through LDS
//   fragment-by-fragment so accumulator pressure shrinks as it goes (no
//   scratch spills by construction).
// ---------------------------------------------------------------------------

typedef __attribute__((ext_vector_type(16))) __bf16 bf16x16;
typedef __attribute__((ext_vector_type(8)))  __bf16 bf16x8;
typedef __attribute__((ext_vector_type(8)))  float  f32x8;

union Frag16 { bf16x16 v; bf16x8 h[2]; };

__device__ __forceinline__ float silu_f(float x) {
  return x * (1.0f / (1.0f + __expf(-x)));
}

// Cox-de Boor, K=3, NUM=5, uniform knots t_j = (j-3)*0.4 - 1, j=0..11.
__device__ __forceinline__ void bspline8(float x, float* B8) {
  const float h = 0.4f;
  float B[11];
#pragma unroll
  for (int j = 0; j < 11; ++j) {
    float t0 = (float)(j - 3) * h - 1.0f;
    float t1 = (float)(j - 2) * h - 1.0f;
    B[j] = (x >= t0 && x < t1) ? 1.0f : 0.0f;
  }
#pragma unroll
  for (int p = 1; p <= 3; ++p) {
    float inv = 1.0f / ((float)p * h);
#pragma unroll
    for (int j = 0; j < 11 - p; ++j) {
      float tj    = (float)(j - 3) * h - 1.0f;      // t[j]
      float tjp1p = (float)(j + p - 2) * h - 1.0f;  // t[j+p+1]
      B[j] = (x - tj) * inv * B[j] + (tjp1p - x) * inv * B[j + 1];
    }
  }
#pragma unroll
  for (int j = 0; j < 8; ++j) B8[j] = B[j];
}

// ---------------------------------------------------------------------------
// F[b][i*8+c] = basis_c(x[b,i]) ; F[b][d*8+i] = silu(x[b,i])   (bf16)
// ---------------------------------------------------------------------------
__global__ void kan_featurize_x(const float* __restrict__ x,
                                __bf16* __restrict__ F, int d, int Kdim) {
  size_t tid = (size_t)blockIdx.x * blockDim.x + threadIdx.x;
  int row = (int)(tid / (size_t)d);
  int i   = (int)(tid - (size_t)row * d);
  float xv = x[tid];
  float B8[8];
  bspline8(xv, B8);
  bf16x8 o;
#pragma unroll
  for (int c = 0; c < 8; ++c) o[c] = (__bf16)B8[c];
  *(bf16x8*)(F + (size_t)row * Kdim + (size_t)i * 8) = o;
  F[(size_t)row * Kdim + (size_t)d * 8 + i] = (__bf16)silu_f(xv);
}

// ---------------------------------------------------------------------------
// Pack weights into WMMA B-fragment order: flat idx = ((nt*KT+kt)*32+lane)*16+e
// B 32x16 bf16 layout: lane half h = lane>>4 holds K = kt*32 + h*16 + e,
// column n = nt*16 + (lane&15).
// virtual K: k < d*8 -> coef[o][k/8][k%8]*sp*mask ; else -> sb[o][k-d*8]*mask
// ---------------------------------------------------------------------------
__global__ void kan_pack_w(const float* __restrict__ coef,
                           const float* __restrict__ sb,
                           const float* __restrict__ sp,
                           const float* __restrict__ mask,
                           __bf16* __restrict__ Wp, int d, int KT) {
  size_t tid  = (size_t)blockIdx.x * blockDim.x + threadIdx.x;
  int e    = (int)(tid & 15);
  int lane = (int)((tid >> 4) & 31);
  size_t blk = tid >> 9;                 // nt*KT + kt
  int kt = (int)(blk % (size_t)KT);
  int nt = (int)(blk / (size_t)KT);
  int hh = lane >> 4;
  int n  = nt * 16 + (lane & 15);
  int k  = kt * 32 + hh * 16 + e;
  int d8 = d * 8;
  float w;
  if (k < d8) {
    int i = k >> 3, c = k & 7;
    size_t ni = (size_t)n * d + i;
    w = coef[ni * 8 + c] * sp[ni] * mask[ni];
  } else {
    size_t ni = (size_t)n * d + (k - d8);
    w = sb[ni] * mask[ni];
  }
  Wp[tid] = (__bf16)w;
}

// ---------------------------------------------------------------------------
// GEMM: C[256 x 128] per workgroup, 8 waves, each wave = 32 rows x 128 cols
// (2 A-frags x 8 B-frags = 16 v_wmma_f32_16x16x32_bf16 per k-tile).
// B tile (8 KB) double-buffered in LDS via async-to-LDS; A frags from L2.
// FEAT=true: epilogue stages C in LDS per fragment, then featurizes.
// ---------------------------------------------------------------------------
#define CS_STRIDE 19                      // LDS f32 row stride (bank-safe)

template <bool FEAT>
__global__ void __launch_bounds__(256, 1)
kan_gemm(const __bf16* __restrict__ A,    // [Mrows x Kdim] row-major bf16
         const __bf16* __restrict__ Bp,   // packed fragment-major weights
         const float* __restrict__ bias,  // [N]
         int Kdim, int KT, int N,
         float* __restrict__ outF32,      // FEAT=false: [Mrows x N]
         __bf16* __restrict__ outFeat,    // FEAT=true : [Mrows x KdimNext]
         int KdimNext) {
  __shared__ union {
    __bf16 bt[2][4096];                   // 2 x 8KB B tiles (GEMM phase)
    float  cs[256 * CS_STRIDE];           // 256x16 C chunk (epilogue phase)
  } sh;

  const int tid  = threadIdx.x;
  const int lane = tid & 31;
  const int wave = tid >> 5;
  const int hh   = lane >> 4;             // K-half selector
  const int ml   = lane & 15;
  const int ntBase = blockIdx.y * 8;
  const size_t rowBase = (size_t)blockIdx.x * 256 + wave * 32;
  const __bf16* aRow0 = A + (rowBase + ml) * (size_t)Kdim;
  const __bf16* aRow1 = A + (rowBase + 16 + ml) * (size_t)Kdim;

  f32x8 acc0[8] = {};
  f32x8 acc1[8] = {};

  // Issue this thread's 2 async 16B chunks of the kt B-tile into LDS buf.
  auto issueB = [&](int kt, int buf) {
#pragma unroll
    for (int j = 0; j < 2; ++j) {
      int chunk = tid * 2 + j;            // 512 chunks x 16B = 8KB
      int f = chunk >> 6;                 // fragment 0..7
      int c = chunk & 63;                 // 16B chunk within fragment
      const __bf16* src =
          Bp + ((size_t)(ntBase + f) * KT + kt) * 512 + (size_t)c * 8;
      unsigned dst = (unsigned)(uintptr_t)(&sh.bt[buf][chunk * 8]);
      asm volatile("global_load_async_to_lds_b128 %0, %1, off"
                   :: "v"(dst), "v"(src) : "memory");
    }
  };

  // A fragment: lane holds row ml, K = 16h..16h+15 (contiguous)
  auto loadA = [&](const __bf16* aRow, int kt) {
    Frag16 a;
    const __bf16* ap = aRow + (size_t)kt * 32 + hh * 16;
    a.h[0] = *(const bf16x8*)(ap);
    a.h[1] = *(const bf16x8*)(ap + 8);
    return a;
  };

  auto compute = [&](int kt, const Frag16& a0, const Frag16& a1) {
    const __bf16* bbase = &sh.bt[kt & 1][0];
#pragma unroll
    for (int f = 0; f < 8; ++f) {
      Frag16 b;
      const __bf16* bp = bbase + f * 512 + lane * 16;   // 32B/lane contiguous
      b.h[0] = *(const bf16x8*)(bp);
      b.h[1] = *(const bf16x8*)(bp + 8);
      acc0[f] = __builtin_amdgcn_wmma_f32_16x16x32_bf16(
          false, a0.v, false, b.v, (short)0, acc0[f], false, false);
      acc1[f] = __builtin_amdgcn_wmma_f32_16x16x32_bf16(
          false, a1.v, false, b.v, (short)0, acc1[f], false, false);
    }
  };

  issueB(0, 0);
  int kt = 0;
  for (; kt < KT - 1; ++kt) {             // branch-free steady state
    Frag16 a0 = loadA(aRow0, kt);
    Frag16 a1 = loadA(aRow1, kt);
    issueB(kt + 1, (kt + 1) & 1);
    asm volatile("s_wait_asynccnt 0x2" ::: "memory");  // tile kt landed
    __syncthreads();
    compute(kt, a0, a1);
    __syncthreads();
  }
  {                                       // peeled last iteration
    Frag16 a0 = loadA(aRow0, kt);
    Frag16 a1 = loadA(aRow1, kt);
    asm volatile("s_wait_asynccnt 0x0" ::: "memory");
    __syncthreads();
    compute(kt, a0, a1);
  }

  // Epilogue. C layout: VGPR r -> M = r + 8*hh, N = ml within each 16x16 tile.
  if (!FEAT) {
    // Direct store: val = acc + bias.
#pragma unroll
    for (int f = 0; f < 8; ++f) {
      int col = (ntBase + f) * 16 + ml;
      float bcol = bias[col];
#pragma unroll
      for (int m = 0; m < 2; ++m) {
        const f32x8& acc = m ? acc1[f] : acc0[f];
#pragma unroll
        for (int r = 0; r < 8; ++r) {
          size_t orow = rowBase + m * 16 + (r + 8 * hh);
          outF32[orow * (size_t)N + col] = acc[r] + bcol;
        }
      }
    }
  } else {
    // Stage each 256x16 C fragment chunk through LDS, then featurize with a
    // rolled loop: bounded temps + accs freed fragment by fragment.
#pragma unroll
    for (int f = 0; f < 8; ++f) {
      float bcol = bias[(ntBase + f) * 16 + ml];
      __syncthreads();                    // chunk buffer free (B-tiles dead /
                                          // previous chunk fully consumed)
#pragma unroll
      for (int m = 0; m < 2; ++m) {
        const f32x8& acc = m ? acc1[f] : acc0[f];
#pragma unroll
        for (int r = 0; r < 8; ++r) {
          int lrow = wave * 32 + m * 16 + (r + 8 * hh);
          sh.cs[lrow * CS_STRIDE + ml] = acc[r] + bcol;
        }
      }
      __syncthreads();
      // 256 threads = 16 rows x 16 cols per pass; 16 passes cover 256 rows.
      const int colL = tid & 15;
      const int row0 = tid >> 4;          // 0..15
      const int colG = (ntBase + f) * 16 + colL;
#pragma unroll 1
      for (int p = 0; p < 16; ++p) {
        int lrow = p * 16 + row0;
        float val = sh.cs[lrow * CS_STRIDE + colL];
        size_t orow = (size_t)blockIdx.x * 256 + lrow;
        float B8[8];
        bspline8(val, B8);
        bf16x8 o;
#pragma unroll
        for (int c = 0; c < 8; ++c) o[c] = (__bf16)B8[c];
        *(bf16x8*)(outFeat + orow * (size_t)KdimNext + (size_t)colG * 8) = o;
        outFeat[orow * (size_t)KdimNext + (size_t)N * 8 + colG] =
            (__bf16)silu_f(val);
      }
    }
  }
}

// ---------------------------------------------------------------------------
extern "C" void kernel_launch(void* const* d_in, const int* in_sizes, int n_in,
                              void* d_out, int out_size, void* d_ws,
                              size_t ws_size, hipStream_t stream) {
  (void)in_sizes; (void)n_in; (void)out_size; (void)ws_size;

  const float* x     = (const float*)d_in[0];
  const float* coef0 = (const float*)d_in[1];
  const float* sb0   = (const float*)d_in[2];
  const float* sp0   = (const float*)d_in[3];
  const float* mask0 = (const float*)d_in[4];
  const float* bias0 = (const float*)d_in[5];
  const float* coef1 = (const float*)d_in[6];
  const float* sb1   = (const float*)d_in[7];
  const float* sp1   = (const float*)d_in[8];
  const float* mask1 = (const float*)d_in[9];
  const float* bias1 = (const float*)d_in[10];

  const int Bch = 8192, d0 = 1024, d1 = 2048, d2 = 1024;
  const int K0 = d0 * 9;            // 9216  (288 k-tiles of 32)
  const int K1 = d1 * 9;            // 18432 (576 k-tiles)
  const int KT0 = K0 / 32, KT1 = K1 / 32;

  // Workspace layout (bf16):
  //   F0 : Bch*K0 = 151.0 MB | W0p: K0*d1 = 37.7 MB
  //   F1 : Bch*K1 = 302.0 MB | W1p: K1*d2 = 37.7 MB
  char* ws = (char*)d_ws;
  __bf16* F0  = (__bf16*)ws;
  __bf16* W0p = (__bf16*)(ws + (size_t)Bch * K0 * 2);
  __bf16* F1  = (__bf16*)(ws + (size_t)Bch * K0 * 2 + (size_t)K0 * d1 * 2);
  __bf16* W1p = (__bf16*)(ws + (size_t)Bch * K0 * 2 + (size_t)K0 * d1 * 2 +
                          (size_t)Bch * K1 * 2);

  // 1) featurize x
  kan_featurize_x<<<(Bch * d0) / 256, 256, 0, stream>>>(x, F0, d0, K0);
  // 2) pack weights (fragment-major bf16)
  kan_pack_w<<<(d1 / 16) * KT0 * 2, 256, 0, stream>>>(coef0, sb0, sp0, mask0,
                                                      W0p, d0, KT0);
  kan_pack_w<<<(d2 / 16) * KT1 * 2, 256, 0, stream>>>(coef1, sb1, sp1, mask1,
                                                      W1p, d1, KT1);
  // 3) layer 0 GEMM, epilogue emits layer-1 features
  dim3 g0(Bch / 256, d1 / 128);
  kan_gemm<true><<<g0, 256, 0, stream>>>(F0, W0p, bias0, K0, KT0, d1, nullptr,
                                         F1, K1);
  // 4) layer 1 GEMM -> f32 output
  dim3 g1(Bch / 256, d2 / 128);
  kan_gemm<false><<<g1, 256, 0, stream>>>(F1, W1p, bias1, K1, KT1, d2,
                                          (float*)d_out, nullptr, 0);
}
